// Mamba3_33371895890360
// MI455X (gfx1250) — matrix operand order repaired and checked by reading, not compiled, not run on hardware
//
#include <hip/hip_runtime.h>
#include <hip/hip_bf16.h>
#include <math.h>

typedef __attribute__((ext_vector_type(16))) _Float16 v16h;
typedef __attribute__((ext_vector_type(8)))  _Float16 v8h;
typedef __attribute__((ext_vector_type(8)))  float    v8f;

#define CAT16(lo, hi) __builtin_shufflevector((lo), (hi), 0,1,2,3,4,5,6,7,8,9,10,11,12,13,14,15)

#define BB   2
#define HH   48
#define WW   48
#define W2   96
#define LL   4608      // HH*W2
#define DIN  128       // d_inner
#define DMOD 64
#define NST  16        // d_state
#define KK4  4
#define NPROJ 48       // dt_rank + 2*d_state = 36, padded to 48

__device__ __forceinline__ float silu_f(float x) { return x / (1.0f + __expf(-x)); }

// ---------------------------------------------------------------------------
// f16 WMMA GEMM, NT independent accumulators per wave (16 x 16*NT C tile).
// All NT B fragments are preloaded before the WMMA group so the compiler can
// clause the loads (one s_wait) and issue NT back-to-back pipelined WMMAs.
// One wave (32 threads) per tile. Kd multiple of 32.
// ---------------------------------------------------------------------------
template <int NT>
__global__ void mamba_gemm_wmma(const _Float16* __restrict__ A,
                                const _Float16* __restrict__ Bw,
                                float* __restrict__ C,
                                int Kd, int lda, int ldb, int ldc)
{
    const int lane = threadIdx.x & 31;
    const int mr   = lane & 15;
    const int grp  = lane >> 4;
    const int m0   = blockIdx.x * 16;
    const int n0   = blockIdx.y * (16 * NT);

    v8f acc[NT];
#pragma unroll
    for (int j = 0; j < NT; ++j) acc[j] = (v8f){};

    const _Float16* arow = A + (size_t)(m0 + mr) * lda + grp * 8;
    for (int kk = 0; kk < Kd; kk += 32) {
        v8h a_lo = *(const v8h*)(arow + kk);
        v8h a_hi = *(const v8h*)(arow + kk + 16);
        v16h av  = CAT16(a_lo, a_hi);
        const _Float16* brow = Bw + (size_t)kk * ldb + n0 + mr;
        v8h bl[NT], bh[NT];
#pragma unroll
        for (int j = 0; j < NT; ++j) {
#pragma unroll
            for (int s = 0; s < 8; ++s) {
                bl[j][s] = brow[(size_t)(grp * 8 + s) * ldb + j * 16];
                bh[j][s] = brow[(size_t)(16 + grp * 8 + s) * ldb + j * 16];
            }
        }
#pragma unroll
        for (int j = 0; j < NT; ++j) {
            v16h bv = CAT16(bl[j], bh[j]);
            acc[j] = __builtin_amdgcn_wmma_f32_16x16x32_f16(false, av, false, bv,
                                                            (short)0, acc[j], false, false);
        }
    }
#pragma unroll
    for (int j = 0; j < NT; ++j)
#pragma unroll
        for (int r = 0; r < 8; ++r)
            C[(size_t)(m0 + r + 8 * grp) * ldc + n0 + j * 16 + mr] = acc[j][r];
}

// ---------------------------------------------------------------------------
// Implicit-GEMM 3x3 conv (pad 1) + bias + BN + ReLU. All 64 out-channels per
// wave (4 independent accumulators; A fragment reused x4, B fragments
// preloaded per k-step). Writes shuffled channels of final NCHW output.
// inNHWC: f16 [B][48][48][128], w9: [9][128][64]
// ---------------------------------------------------------------------------
__global__ void mamba_conv3x3_wmma(const _Float16* __restrict__ inNHWC,
                                   const _Float16* __restrict__ w9,
                                   const float* __restrict__ bias,
                                   const float* __restrict__ bng,
                                   const float* __restrict__ bnb,
                                   float* __restrict__ outp,
                                   int chanOff)
{
    const int lane = threadIdx.x & 31;
    const int mr   = lane & 15;
    const int grp  = lane >> 4;
    const int mt   = blockIdx.x;            // 0..287 (B * 2304/16)
    const int b    = mt / 144;
    const int sp0  = (mt % 144) * 16;
    const int spA  = sp0 + mr;
    const int ya   = spA / WW, xa = spA % WW;

    v8f acc[4];
#pragma unroll
    for (int j = 0; j < 4; ++j) acc[j] = (v8f){};

    for (int tap = 0; tap < 9; ++tap) {
        const int dy = tap / 3 - 1, dx = tap % 3 - 1;
        const int ys = ya + dy, xs = xa + dx;
        const bool inb = (ys >= 0 && ys < HH && xs >= 0 && xs < WW);
        const _Float16* arow = inNHWC + (((size_t)(b * HH + ys) * WW + xs) << 7);
        for (int cc = 0; cc < DIN; cc += 32) {
            v8h a_lo = {}, a_hi = {};
            if (inb) {
                a_lo = *(const v8h*)(arow + cc + grp * 8);
                a_hi = *(const v8h*)(arow + cc + grp * 8 + 16);
            }
            v16h av = CAT16(a_lo, a_hi);
            const _Float16* brow = w9 + (size_t)(tap * DIN + cc) * 64 + mr;
            v8h bl[4], bh[4];
#pragma unroll
            for (int j = 0; j < 4; ++j) {
#pragma unroll
                for (int s = 0; s < 8; ++s) {
                    bl[j][s] = brow[(grp * 8 + s) * 64 + j * 16];
                    bh[j][s] = brow[(16 + grp * 8 + s) * 64 + j * 16];
                }
            }
#pragma unroll
            for (int j = 0; j < 4; ++j) {
                v16h bv = CAT16(bl[j], bh[j]);
                acc[j] = __builtin_amdgcn_wmma_f32_16x16x32_f16(false, av, false, bv,
                                                                (short)0, acc[j], false, false);
            }
        }
    }
#pragma unroll
    for (int j = 0; j < 4; ++j) {
        const int   o     = j * 16 + mr;
        const float scale = bng[o] * rsqrtf(1.0f + 1e-5f);
        const float bs    = bias[o];
        const float bb    = bnb[o];
        const int   jc    = 2 * o + chanOff;   // group-shuffle (g=2) folded in
#pragma unroll
        for (int r = 0; r < 8; ++r) {
            const int m  = sp0 + r + 8 * grp;
            const int yy = m / WW, xx = m % WW;
            float v = (acc[j][r] + bs) * scale + bb;
            v = v > 0.0f ? v : 0.0f;
            outp[(((size_t)b * 128 + jc) * HH + yy) * WW + xx] = v;
        }
    }
}

// ---------------------------------------------------------------------------
// Staging / packing kernels (f32 -> f16, layout transforms)
// ---------------------------------------------------------------------------
__global__ void mamba_stage_conv1_in(const float* __restrict__ rgb,
                                     const float* __restrict__ t,
                                     _Float16* __restrict__ nhwc)
{
    int idx = blockIdx.x * blockDim.x + threadIdx.x;   // B*48*48*128
    if (idx >= BB * HH * WW * DIN) return;
    int c = idx & 127; int p = idx >> 7;
    int x = p % WW; p /= WW; int y = p % HH; int b = p / HH;
    float v = (c < 64) ? rgb[((size_t)(b * 128 + c)) * 2304 + y * WW + x]
                       : t[((size_t)(b * 128 + (c - 64))) * 2304 + y * WW + x];
    nhwc[idx] = (_Float16)v;
}

__global__ void mamba_stage_rgbtA(const float* __restrict__ rgb,
                                  const float* __restrict__ t,
                                  _Float16* __restrict__ outA)   // [9216][64]
{
    int idx = blockIdx.x * blockDim.x + threadIdx.x;
    if (idx >= BB * LL * DMOD) return;
    int c = idx & 63; int m = idx >> 6;
    int b = m / LL, l = m % LL;
    int h = l / W2, w2 = l % W2;
    int x = w2 >> 1;
    const float* src = (w2 & 1) ? t : rgb;
    outA[idx] = (_Float16)src[((size_t)(b * 128 + 64 + c)) * 2304 + h * WW + x];
}

__global__ void mamba_pack_convw(const float* __restrict__ w,   // [64][128][3][3]
                                 _Float16* __restrict__ w9)     // [9][128][64]
{
    int idx = blockIdx.x * blockDim.x + threadIdx.x;
    if (idx >= 9 * DIN * 64) return;
    int o = idx & 63; int tcc = idx >> 6;
    int c = tcc % DIN, tap = tcc / DIN;
    w9[idx] = (_Float16)w[((size_t)o * DIN + c) * 9 + tap];
}

__global__ void mamba_pack_ipB(const float* __restrict__ w,   // [256][64]
                               _Float16* __restrict__ o)      // [64][256]
{
    int idx = blockIdx.x * blockDim.x + threadIdx.x;
    if (idx >= 64 * 256) return;
    int e = idx & 255, c = idx >> 8;
    o[idx] = (_Float16)w[(size_t)e * 64 + c];
}

__global__ void mamba_pack_xpB(const float* __restrict__ w,   // [4][36][128]
                               _Float16* __restrict__ o)      // [4][128][48]
{
    int idx = blockIdx.x * blockDim.x + threadIdx.x;
    if (idx >= KK4 * DIN * NPROJ) return;
    int c = idx % NPROJ; int kd = idx / NPROJ;
    int d = kd % DIN, k = kd / DIN;
    o[idx] = (c < 36) ? (_Float16)w[((size_t)k * 36 + c) * DIN + d] : (_Float16)0.0f;
}

__global__ void mamba_pack_opB(const float* __restrict__ w,   // [64][128]
                               _Float16* __restrict__ o)      // [128][64]
{
    int idx = blockIdx.x * blockDim.x + threadIdx.x;
    if (idx >= DIN * 64) return;
    int c = idx & 63, d = idx >> 6;
    o[idx] = (_Float16)w[(size_t)c * DIN + d];
}

// ---------------------------------------------------------------------------
// Depthwise 3x3 + SiLU on in-proj "x" half. xz: [B*L][256], xd: [B*L][128]
// ---------------------------------------------------------------------------
__global__ void mamba_dwconv_silu(const float* __restrict__ xz,
                                  const float* __restrict__ w,    // [128][9]
                                  const float* __restrict__ bias,
                                  float* __restrict__ xd)
{
    int idx = blockIdx.x * blockDim.x + threadIdx.x;
    if (idx >= BB * LL * DIN) return;
    int c = idx & 127; int p = idx >> 7;
    int x = p % W2; p /= W2; int y = p % HH; int b = p / HH;
    float s = bias[c];
#pragma unroll
    for (int tap = 0; tap < 9; ++tap) {
        int dy = tap / 3 - 1, dx = tap % 3 - 1;
        int ys = y + dy, xs = x + dx;
        if (ys >= 0 && ys < HH && xs >= 0 && xs < W2)
            s += w[c * 9 + tap] * xz[((size_t)(b * LL + ys * W2 + xs)) * 256 + c];
    }
    xd[idx] = silu_f(s);
}

// xs directions staged as f16 [B][4][L][128] (scan-index major, d contiguous)
__global__ void mamba_stage_xs(const float* __restrict__ xd,
                               _Float16* __restrict__ xsA)
{
    int idx = blockIdx.x * blockDim.x + threadIdx.x;
    if (idx >= BB * KK4 * LL * DIN) return;
    int d = idx & 127; int t = idx >> 7;
    int l = t % LL; t /= LL;
    int k = t & 3; int b = t >> 2;
    int src;
    if (k == 0)      src = l;
    else if (k == 1) { int w = l / HH, h = l % HH; src = h * W2 + w; }
    else if (k == 2) src = LL - 1 - l;
    else             { int l2 = LL - 1 - l; int w = l2 / HH, h = l2 % HH; src = h * W2 + w; }
    xsA[idx] = (_Float16)xd[((size_t)(b * LL + src)) * DIN + d];
}

// delta[b][k][d][l] = softplus(dt_b + sum_r x_dbl[b][k][l][r]*dt_w[k][d][r])
__global__ void mamba_delta(const float* __restrict__ xdbl,   // [B][4][L][48]
                            const float* __restrict__ dtw,    // [4][128][4]
                            const float* __restrict__ dtb,    // [4][128]
                            float* __restrict__ delta)
{
    int idx = blockIdx.x * blockDim.x + threadIdx.x;
    if (idx >= BB * KK4 * DIN * LL) return;
    int l = idx % LL; int t = idx / LL;
    int d = t & 127; t >>= 7;
    int k = t & 3; int b = t >> 2;
    const float* row = xdbl + ((size_t)((b * 4 + k) * LL) + l) * NPROJ;
    float s = dtb[k * DIN + d];
#pragma unroll
    for (int r = 0; r < 4; ++r) s += row[r] * dtw[(k * DIN + d) * 4 + r];
    delta[idx] = (s > 20.0f) ? s : __logf(1.0f + __expf(s));
}

// ---------------------------------------------------------------------------
// Selective scan: 16 lanes per (b,k,d) sequence, lane n owns state n.
// ---------------------------------------------------------------------------
__global__ void mamba_scan(const _Float16* __restrict__ xsA,   // [B][4][L][128]
                           const float* __restrict__ xdbl,     // [B][4][L][48]
                           const float* __restrict__ delta,    // [B][4][128][L]
                           const float* __restrict__ A_logs,   // [4][128][16]
                           const float* __restrict__ Ds,       // [4][128]
                           float* __restrict__ out_y)          // [B][4][128][L]
{
    const int gid = blockIdx.x * blockDim.x + threadIdx.x;
    const int n   = gid & 15;
    const int seq = gid >> 4;            // 0..1023
    const int d   = seq & 127;
    const int k   = (seq >> 7) & 3;
    const int b   = seq >> 9;

    const float Aval = -__expf(A_logs[(k * DIN + d) * NST + n]);
    const float Dval = Ds[k * DIN + d];
    const _Float16* up  = xsA  + ((size_t)(b * 4 + k) * LL) * DIN + d;
    const float* xrow   = xdbl + ((size_t)(b * 4 + k) * LL) * NPROJ;
    const float* dtp    = delta + ((size_t)((b * 4 + k) * DIN + d)) * LL;
    float* yp           = out_y + ((size_t)((b * 4 + k) * DIN + d)) * LL;

    float h = 0.0f;
    for (int l = 0; l < LL; ++l) {
        __builtin_prefetch(xrow + (size_t)(l + 24 < LL ? l + 24 : LL - 1) * NPROJ, 0, 1);
        const float u  = (float)up[(size_t)l * DIN];
        const float dt = dtp[l];
        const float Bn = xrow[(size_t)l * NPROJ + 4 + n];
        const float Cn = xrow[(size_t)l * NPROJ + 20 + n];
        const float dA = __expf(dt * Aval);
        h = dA * h + (dt * u) * Bn;
        float v = h * Cn;
        v += __shfl_xor(v, 1, 32);
        v += __shfl_xor(v, 2, 32);
        v += __shfl_xor(v, 4, 32);
        v += __shfl_xor(v, 8, 32);
        if (n == 0) yp[l] = v + Dval * u;
    }
}

// ---------------------------------------------------------------------------
// Combine 4 directions (with index remaps) + layernorm(d) + SiLU gate -> f16 A
// ---------------------------------------------------------------------------
__global__ void mamba_combine_ln_gate(const float* __restrict__ out_y,
                                      const float* __restrict__ xz,
                                      const float* __restrict__ g,
                                      const float* __restrict__ beta,
                                      _Float16* __restrict__ yA)
{
    const int bl = blockIdx.x;           // 0..9215
    const int b  = bl / LL;
    const int l  = bl % LL;
    const int d  = threadIdx.x;          // 0..127
    const int h  = l / W2, w = l % W2;
    const int l2 = w * HH + h;
    const size_t base = (size_t)(b * 4) * DIN * LL;
    float y = out_y[base + ((size_t)(0 * DIN + d)) * LL + l]
            + out_y[base + ((size_t)(1 * DIN + d)) * LL + l2]
            + out_y[base + ((size_t)(2 * DIN + d)) * LL + (LL - 1 - l)]
            + out_y[base + ((size_t)(3 * DIN + d)) * LL + (LL - 1 - l2)];

    __shared__ float red[DIN];
    red[d] = y;
    __syncthreads();
    for (int s = 64; s > 0; s >>= 1) { if (d < s) red[d] += red[d + s]; __syncthreads(); }
    const float mean = red[0] * (1.0f / DIN);
    __syncthreads();
    const float cdev = y - mean;
    red[d] = cdev * cdev;
    __syncthreads();
    for (int s = 64; s > 0; s >>= 1) { if (d < s) red[d] += red[d + s]; __syncthreads(); }
    const float var = red[0] * (1.0f / DIN);

    const float ln = cdev * rsqrtf(var + 1e-5f) * g[d] + beta[d];
    const float z  = xz[((size_t)bl) * 256 + 128 + d];
    yA[(size_t)bl * DIN + d] = (_Float16)(ln * silu_f(z));
}

// conv2 input staging: de-interleave ss2d output, add residuals, -> NHWC f16
__global__ void mamba_stage_conv2_in(const float* __restrict__ outC,  // [B*L][64]
                                     const float* __restrict__ rgb,
                                     const float* __restrict__ t,
                                     _Float16* __restrict__ nhwc)
{
    int idx = blockIdx.x * blockDim.x + threadIdx.x;
    if (idx >= BB * HH * WW * DIN) return;
    int c = idx & 127; int p = idx >> 7;
    int x = p % WW; p /= WW; int y = p % HH; int b = p / HH;
    float v;
    if (c < 64) {
        v = outC[((size_t)(b * LL + y * W2 + 2 * x)) * 64 + c]
          + rgb[((size_t)(b * 128 + 64 + c)) * 2304 + y * WW + x];
    } else {
        int cc = c - 64;
        v = outC[((size_t)(b * LL + y * W2 + 2 * x + 1)) * 64 + cc]
          + t[((size_t)(b * 128 + 64 + cc)) * 2304 + y * WW + x];
    }
    nhwc[idx] = (_Float16)v;
}

// ---------------------------------------------------------------------------
extern "C" void kernel_launch(void* const* d_in, const int* in_sizes, int n_in,
                              void* d_out, int out_size, void* d_ws, size_t ws_size,
                              hipStream_t stream)
{
    (void)in_sizes; (void)n_in; (void)out_size; (void)ws_size;
    const float* rgb       = (const float*)d_in[0];
    const float* t_in      = (const float*)d_in[1];
    const float* conv1_w   = (const float*)d_in[2];
    const float* conv1_b   = (const float*)d_in[3];
    const float* bn1_g     = (const float*)d_in[4];
    const float* bn1_b     = (const float*)d_in[5];
    const float* conv2_w   = (const float*)d_in[6];
    const float* conv2_b   = (const float*)d_in[7];
    const float* bn2_g     = (const float*)d_in[8];
    const float* bn2_b     = (const float*)d_in[9];
    const float* in_proj_w = (const float*)d_in[10];
    const float* conv_dw_w = (const float*)d_in[11];
    const float* conv_dw_b = (const float*)d_in[12];
    const float* x_proj_w  = (const float*)d_in[13];
    const float* dt_w      = (const float*)d_in[14];
    const float* dt_b      = (const float*)d_in[15];
    const float* A_logs    = (const float*)d_in[16];
    const float* Ds        = (const float*)d_in[17];
    const float* on_g      = (const float*)d_in[18];
    const float* on_b      = (const float*)d_in[19];
    const float* out_proj_w= (const float*)d_in[20];
    float* outp = (float*)d_out;

    char* ws = (char*)d_ws;
    size_t off = 0;
    auto alloc = [&](size_t bytes) -> void* {
        void* p = ws + off;
        off += (bytes + 255) & ~(size_t)255;
        return p;
    };
    _Float16* nhwc1 = (_Float16*)alloc((size_t)589824 * 2);
    _Float16* w9a   = (_Float16*)alloc((size_t)73728 * 2);
    _Float16* w9b   = (_Float16*)alloc((size_t)73728 * 2);
    _Float16* ipB   = (_Float16*)alloc((size_t)16384 * 2);
    _Float16* xpB   = (_Float16*)alloc((size_t)24576 * 2);
    _Float16* opB   = (_Float16*)alloc((size_t)8192 * 2);
    _Float16* rgbtA = (_Float16*)alloc((size_t)589824 * 2);
    float*    xz    = (float*)alloc((size_t)2359296 * 4);
    float*    xd    = (float*)alloc((size_t)1179648 * 4);
    _Float16* xsA   = (_Float16*)alloc((size_t)4718592 * 2);
    float*    xdbl  = (float*)alloc((size_t)1769472 * 4);
    float*    dlt   = (float*)alloc((size_t)4718592 * 4);
    float*    oy    = (float*)alloc((size_t)4718592 * 4);
    _Float16* yA    = (_Float16*)alloc((size_t)1179648 * 2);
    float*    outC  = (float*)alloc((size_t)589824 * 4);
    _Float16* nhwc2 = (_Float16*)alloc((size_t)589824 * 2);

    // Stage weights + inputs
    mamba_stage_conv1_in<<<(589824 + 255) / 256, 256, 0, stream>>>(rgb, t_in, nhwc1);
    mamba_pack_convw<<<(73728 + 255) / 256, 256, 0, stream>>>(conv1_w, w9a);
    mamba_pack_convw<<<(73728 + 255) / 256, 256, 0, stream>>>(conv2_w, w9b);
    mamba_pack_ipB<<<(16384 + 255) / 256, 256, 0, stream>>>(in_proj_w, ipB);
    mamba_pack_xpB<<<(24576 + 255) / 256, 256, 0, stream>>>(x_proj_w, xpB);
    mamba_pack_opB<<<(8192 + 255) / 256, 256, 0, stream>>>(out_proj_w, opB);
    mamba_stage_rgbtA<<<(589824 + 255) / 256, 256, 0, stream>>>(rgb, t_in, rgbtA);

    // in-proj: [9216,64] x [64,256] -> xz   (4 N-tiles per wave)
    mamba_gemm_wmma<4><<<dim3(576, 4), 32, 0, stream>>>(rgbtA, ipB, xz, 64, 64, 256, 256);

    // depthwise conv + silu, then 4-direction staging
    mamba_dwconv_silu<<<(1179648 + 255) / 256, 256, 0, stream>>>(xz, conv_dw_w, conv_dw_b, xd);
    mamba_stage_xs<<<(4718592 + 255) / 256, 256, 0, stream>>>(xd, xsA);

    // x-proj per (b,k): [4608,128] x [128,48] -> x_dbl  (3 N-tiles per wave)
    for (int b = 0; b < BB; ++b)
        for (int k = 0; k < KK4; ++k)
            mamba_gemm_wmma<3><<<dim3(288, 1), 32, 0, stream>>>(
                xsA + (size_t)(b * 4 + k) * LL * DIN,
                xpB + (size_t)k * DIN * NPROJ,
                xdbl + (size_t)(b * 4 + k) * LL * NPROJ,
                128, 128, NPROJ, NPROJ);

    mamba_delta<<<(4718592 + 255) / 256, 256, 0, stream>>>(xdbl, dt_w, dt_b, dlt);
    mamba_scan<<<64, 256, 0, stream>>>(xsA, xdbl, dlt, A_logs, Ds, oy);
    mamba_combine_ln_gate<<<9216, 128, 0, stream>>>(oy, xz, on_g, on_b, yA);

    // out-proj: [9216,128] x [128,64] -> outC  (4 N-tiles per wave)
    mamba_gemm_wmma<4><<<dim3(576, 1), 32, 0, stream>>>(yA, opB, outC, 128, 128, 64, 64);

    mamba_stage_conv2_in<<<(589824 + 255) / 256, 256, 0, stream>>>(outC, rgb, t_in, nhwc2);

    // right -> even shuffled channels, left -> odd shuffled channels
    mamba_conv3x3_wmma<<<dim3(288), 32, 0, stream>>>(nhwc2, w9b, conv2_b, bn2_g, bn2_b, outp, 0);
    mamba_conv3x3_wmma<<<dim3(288), 32, 0, stream>>>(nhwc1, w9a, conv1_b, bn1_g, bn1_b, outp, 1);
}
